// HybridGATLayer_16690242912869
// MI455X (gfx1250) — compile-verified, matrix operand order checked by reading, not run on hardware
//
#include <hip/hip_runtime.h>
#include <hip/hip_bf16.h>
#include <math.h>

#define HID 128
#define HEADS 4
#define NEG_SLOPE 0.2f

typedef __attribute__((ext_vector_type(16))) __bf16 v16bf;
typedef __attribute__((ext_vector_type(8)))  __bf16 v8bf;
typedef __attribute__((ext_vector_type(8)))  float  v8f;

static constexpr int SA = 136;   // LDS row stride (halves) for A tile (16 x 128)
static constexpr int SW = 136;   // LDS row stride (halves) for W^T tile (64 x 128)

// ---------------------------------------------------------------------------
// bf16 WMMA GEMM:  out = act( A[rowTile] @ W + bias )
// A: [N,128] (f32 or bf16), W: [128,Nout] f32 row-major.
// Block = 128 threads (4 waves). Each block: 16 rows x 64 cols.
// Each wave: one 16x16 tile, K-loop of 4 x (16x16x32) WMMA.
// bf16-A path stages the A tile with GLOBAL_LOAD_ASYNC_TO_LDS_B128 (ASYNCcnt),
// overlapped with the VALU transpose/convert of the W^T tile.
// ---------------------------------------------------------------------------
template <bool A_BF16, bool RELU, bool HAS_BIAS, bool OUT_F, bool OUT_B>
__global__ __launch_bounds__(128) void gat_gemm_kernel(
    const float* __restrict__ Af, const __bf16* __restrict__ Ab,
    const float* __restrict__ W, const float* __restrict__ bias,
    int Nout, float* __restrict__ outF, __bf16* __restrict__ outB)
{
    __shared__ __align__(16) __bf16 ldsA[16 * SA];
    __shared__ __align__(16) __bf16 ldsW[64 * SW];

    const int tid = threadIdx.x;
    const int wave = tid >> 5, lane = tid & 31;
    const int rowBase = blockIdx.x * 16;
    const int colBase = blockIdx.y * 64;

    if constexpr (A_BF16) {
        // Async DMA: wave w copies rows 4w..4w+3. Each op: 32 lanes x 16B = 2 rows.
        // LDS byte offset = low 32 bits of the generic shared pointer (flat-LDS
        // aperture).
        const unsigned ldsBase = (unsigned)(size_t)(void*)&ldsA[0];
#pragma unroll
        for (int j = 0; j < 2; ++j) {
            int row = wave * 4 + j * 2 + (lane >> 4);
            unsigned bIn = (unsigned)(lane & 15) * 16u;
            unsigned ldsOff = ldsBase + (unsigned)(row * SA * 2) + bIn;
            unsigned long long ga = (unsigned long long)(size_t)
                ((const char*)(Ab + (size_t)(rowBase + row) * HID) + bIn);
            asm volatile("global_load_async_to_lds_b128 %0, %1, off"
                         :: "v"(ldsOff), "v"(ga) : "memory");
        }
    } else {
        // f32 input: convert while staging. Coalesced: tid -> consecutive K.
#pragma unroll
        for (int i = 0; i < 16; ++i) {
            int idx = i * 128 + tid;
            int r = idx >> 7, k = idx & 127;
            ldsA[r * SA + k] = (__bf16)Af[(size_t)(rowBase + r) * HID + k];
        }
    }

    // Stage W^T tile: ldsW[n][k] = W[k][colBase+n]. Coalesced global reads over n.
    // Overlaps with the async A DMA above.
#pragma unroll 8
    for (int i = 0; i < 64; ++i) {
        int idx = i * 128 + tid;
        int n = idx & 63, k = idx >> 6;
        ldsW[n * SW + k] = (__bf16)W[(size_t)k * Nout + colBase + n];
    }

    if constexpr (A_BF16) {
        asm volatile("s_wait_asynccnt 0x0" ::: "memory");
    }
    __syncthreads();

    const int g = lane >> 4;       // lane group (K split)
    const int mr = lane & 15;      // A row / B column / C column index
    const int n0 = wave * 16;

    v8f c = {};
#pragma unroll
    for (int kt = 0; kt < HID; kt += 32) {
        // A operand: VGPR0-3 = K kt+[0..7] (g0) / kt+[8..15] (g1); VGPR4-7 = +16.
        v8bf a0 = *(const v8bf*)&ldsA[mr * SA + kt + g * 8];
        v8bf a1 = *(const v8bf*)&ldsA[mr * SA + kt + 16 + g * 8];
        // B operand: same striping on W^T rows (lane = output column).
        v8bf b0 = *(const v8bf*)&ldsW[(n0 + mr) * SW + kt + g * 8];
        v8bf b1 = *(const v8bf*)&ldsW[(n0 + mr) * SW + kt + 16 + g * 8];
        v16bf a, b;
#pragma unroll
        for (int i = 0; i < 8; ++i) {
            a[i] = a0[i]; a[i + 8] = a1[i];
            b[i] = b0[i]; b[i + 8] = b1[i];
        }
        c = __builtin_amdgcn_wmma_f32_16x16x32_bf16(
                false, a, false, b, (short)0, c, false, false);
    }

    // C/D layout: element (M = v + 8*g, N = lane&15) in VGPR v.
    const int Ncol = colBase + n0 + mr;
    float bv = 0.0f;
    if constexpr (HAS_BIAS) bv = bias[Ncol];
#pragma unroll
    for (int v = 0; v < 8; ++v) {
        int M = rowBase + v + 8 * g;
        float val = c[v] + bv;
        if constexpr (RELU) { if (val < 0.0f) val = 0.0f; }
        if constexpr (OUT_F) outF[(size_t)M * Nout + Ncol] = val;
        if constexpr (OUT_B) outB[(size_t)M * Nout + Ncol] = (__bf16)val;
    }
}

// ---------------------------------------------------------------------------
__device__ __forceinline__ float waveReduceSum(float v) {
#pragma unroll
    for (int off = 16; off > 0; off >>= 1) v += __shfl_xor(v, off, 32);
    return v;
}

// LayerNorm over 128 channels; one wave per row; writes bf16 for next GEMM.
__global__ __launch_bounds__(256) void gat_ln_kernel(
    const float* __restrict__ h2, const float* __restrict__ g,
    const float* __restrict__ b, __bf16* __restrict__ hb, int N)
{
    int wave = threadIdx.x >> 5, lane = threadIdx.x & 31;
    int r = blockIdx.x * 8 + wave;
    if (r >= N) return;
    float4 v = ((const float4*)(h2 + (size_t)r * HID))[lane];
    float s = waveReduceSum(v.x + v.y + v.z + v.w);
    float mu = s * (1.0f / HID);
    float dx = v.x - mu, dy = v.y - mu, dz = v.z - mu, dw = v.w - mu;
    float s2 = waveReduceSum(dx * dx + dy * dy + dz * dz + dw * dw);
    float rs = rsqrtf(s2 * (1.0f / HID) + 1e-5f);
    int c = lane * 4;
    size_t base = (size_t)r * HID + c;
    hb[base + 0] = (__bf16)(dx * rs * g[c + 0] + b[c + 0]);
    hb[base + 1] = (__bf16)(dy * rs * g[c + 1] + b[c + 1]);
    hb[base + 2] = (__bf16)(dz * rs * g[c + 2] + b[c + 2]);
    hb[base + 3] = (__bf16)(dw * rs * g[c + 3] + b[c + 3]);
}

// a_src[n,h] = dot(xh[n,h,:], att_src[h,:]); a_dst likewise. One wave per node.
__global__ __launch_bounds__(256) void gat_att_kernel(
    const float* __restrict__ xh, const float* __restrict__ att_s,
    const float* __restrict__ att_d, float* __restrict__ a_s,
    float* __restrict__ a_d, int N)
{
    int wave = threadIdx.x >> 5, lane = threadIdx.x & 31;
    int n = blockIdx.x * 8 + wave;
    if (n >= N) return;
#pragma unroll
    for (int h = 0; h < HEADS; ++h) {
        float4 v = ((const float4*)(xh + (size_t)n * (HEADS * HID) + h * HID))[lane];
        float4 s = ((const float4*)(att_s + h * HID))[lane];
        float4 d = ((const float4*)(att_d + h * HID))[lane];
        float ps = waveReduceSum(v.x * s.x + v.y * s.y + v.z * s.z + v.w * s.w);
        float pd = waveReduceSum(v.x * d.x + v.y * d.y + v.z * d.z + v.w * d.w);
        if (lane == 0) { a_s[n * HEADS + h] = ps; a_d[n * HEADS + h] = pd; }
    }
}

__global__ void gat_init_kernel(float* __restrict__ out, float* __restrict__ m,
                                float* __restrict__ den, int total, int nm)
{
    int i = blockIdx.x * 256 + threadIdx.x;
    if (i < total) out[i] = 0.0f;
    if (i < nm) { m[i] = -INFINITY; den[i] = 0.0f; }
}

__device__ __forceinline__ void atomicMaxF(float* addr, float val) {
    if (val >= 0.0f) atomicMax((int*)addr, __float_as_int(val));
    else             atomicMin((unsigned int*)addr, __float_as_uint(val));
}

__device__ __forceinline__ void edgeSrcDst(const long long* ei, int e, int E,
                                           int& s, int& d) {
    if (e < E) { s = (int)ei[e]; d = (int)ei[(size_t)E + e]; }
    else       { s = d = e - E; }   // self loops appended
}

__global__ void gat_edge_max_kernel(const long long* __restrict__ ei,
                                    const float* __restrict__ a_s,
                                    const float* __restrict__ a_d,
                                    float* __restrict__ m, int E, int E2)
{
    int t = blockIdx.x * 256 + threadIdx.x;
    if (t >= E2 * HEADS) return;
    int e = t >> 2, h = t & 3;
    int s, d; edgeSrcDst(ei, e, E, s, d);
    float v = a_s[s * HEADS + h] + a_d[d * HEADS + h];
    v = v > 0.0f ? v : NEG_SLOPE * v;
    atomicMaxF(&m[d * HEADS + h], v);
}

__global__ void gat_edge_sum_kernel(const long long* __restrict__ ei,
                                    const float* __restrict__ a_s,
                                    const float* __restrict__ a_d,
                                    const float* __restrict__ m,
                                    float* __restrict__ den, int E, int E2)
{
    int t = blockIdx.x * 256 + threadIdx.x;
    if (t >= E2 * HEADS) return;
    int e = t >> 2, h = t & 3;
    int s, d; edgeSrcDst(ei, e, E, s, d);
    float v = a_s[s * HEADS + h] + a_d[d * HEADS + h];
    v = v > 0.0f ? v : NEG_SLOPE * v;
    float mv = m[d * HEADS + h];
    if (!(mv > -1e30f)) mv = 0.0f;
    atomicAdd(&den[d * HEADS + h], __expf(v - mv));
}

// Scatter: out[dst,c] += 0.25 * sum_h alpha(e,h) * xh[src,h,c]. 2 edges / block.
// Prefetch the random xh[src] gather lines as soon as src is known so the
// HBM fetch overlaps the alpha computation (this kernel is the roofline
// bottleneck: ~2.2 GB gather+atomic traffic).
__global__ __launch_bounds__(256) void gat_edge_scatter_kernel(
    const long long* __restrict__ ei, const float* __restrict__ a_s,
    const float* __restrict__ a_d, const float* __restrict__ m,
    const float* __restrict__ den, const float* __restrict__ xh,
    float* __restrict__ out, int E, int E2)
{
    int e = blockIdx.x * 2 + (threadIdx.x >> 7);
    int c = threadIdx.x & 127;
    if (e >= E2) return;
    int s, d; edgeSrcDst(ei, e, E, s, d);
    const float* xp = xh + (size_t)s * (HEADS * HID) + c;
#pragma unroll
    for (int h = 0; h < HEADS; ++h)
        __builtin_prefetch(xp + h * HID, 0, 1);   // global_prefetch_b8
    float acc = 0.0f;
#pragma unroll
    for (int h = 0; h < HEADS; ++h) {
        float v = a_s[s * HEADS + h] + a_d[d * HEADS + h];
        v = v > 0.0f ? v : NEG_SLOPE * v;
        float mv = m[d * HEADS + h];
        if (!(mv > -1e30f)) mv = 0.0f;
        float al = __expf(v - mv) / (den[d * HEADS + h] + 1e-16f);
        acc += al * xp[h * HID];
    }
    atomicAdd(&out[(size_t)d * HID + c], 0.25f * acc);
}

__global__ void gat_bias_kernel(float* __restrict__ out,
                                const float* __restrict__ bias, int total)
{
    int i = blockIdx.x * 256 + threadIdx.x;
    if (i < total) out[i] += bias[i & (HID - 1)];
}

// ---------------------------------------------------------------------------
extern "C" void kernel_launch(void* const* d_in, const int* in_sizes, int n_in,
                              void* d_out, int out_size, void* d_ws, size_t ws_size,
                              hipStream_t stream) {
    const float*     x       = (const float*)d_in[0];
    const long long* ei      = (const long long*)d_in[1];   // int64 edge_index [2,E]
    const float*     w1      = (const float*)d_in[2];
    const float*     b1      = (const float*)d_in[3];
    const float*     w2      = (const float*)d_in[4];
    const float*     b2      = (const float*)d_in[5];
    const float*     ln_g    = (const float*)d_in[6];
    const float*     ln_b    = (const float*)d_in[7];
    const float*     w_lin   = (const float*)d_in[8];
    const float*     att_src = (const float*)d_in[9];
    const float*     att_dst = (const float*)d_in[10];
    const float*     bias    = (const float*)d_in[11];

    const int N  = in_sizes[0] / HID;
    const int E  = in_sizes[1] / 2;
    const int E2 = E + N;

    char* ws = (char*)d_ws;
    size_t off = 0;
    auto alloc = [&](size_t bytes) -> void* {
        void* p = ws + off;
        off = (off + bytes + 255) & ~(size_t)255;
        return p;
    };
    __bf16* h1b = (__bf16*)alloc((size_t)N * HID * sizeof(__bf16));
    float*  h2  = (float*) alloc((size_t)N * HID * sizeof(float));
    __bf16* hb  = (__bf16*)alloc((size_t)N * HID * sizeof(__bf16));
    float*  xh  = (float*) alloc((size_t)N * HEADS * HID * sizeof(float));
    float*  a_s = (float*) alloc((size_t)N * HEADS * sizeof(float));
    float*  a_d = (float*) alloc((size_t)N * HEADS * sizeof(float));
    float*  m   = (float*) alloc((size_t)N * HEADS * sizeof(float));
    float*  den = (float*) alloc((size_t)N * HEADS * sizeof(float));
    float*  out = (float*)d_out;

    // MLP: h1 = relu(x@w1+b1)  (f32 A path; bf16 out feeds next GEMM)
    gat_gemm_kernel<false, true, true, false, true>
        <<<dim3(N / 16, HID / 64), 128, 0, stream>>>(
            x, nullptr, w1, b1, HID, nullptr, h1b);
    // h2 = relu(h1@w2+b2)  (async bf16 A; f32 out for layernorm)
    gat_gemm_kernel<true, true, true, true, false>
        <<<dim3(N / 16, HID / 64), 128, 0, stream>>>(
            nullptr, h1b, w2, b2, HID, h2, nullptr);
    // layernorm -> bf16
    gat_ln_kernel<<<N / 8, 256, 0, stream>>>(h2, ln_g, ln_b, hb, N);
    // xh = h @ w_lin  [N, 4*128] f32  (async bf16 A)
    gat_gemm_kernel<true, false, false, true, false>
        <<<dim3(N / 16, (HEADS * HID) / 64), 128, 0, stream>>>(
            nullptr, hb, w_lin, nullptr, HEADS * HID, xh, nullptr);
    // attention coefficients
    gat_att_kernel<<<N / 8, 256, 0, stream>>>(xh, att_src, att_dst, a_s, a_d, N);
    // init out / segment buffers
    gat_init_kernel<<<(N * HID + 255) / 256, 256, 0, stream>>>(
        out, m, den, N * HID, N * HEADS);
    // segment softmax (max, sum) then scatter
    gat_edge_max_kernel<<<(E2 * HEADS + 255) / 256, 256, 0, stream>>>(
        ei, a_s, a_d, m, E, E2);
    gat_edge_sum_kernel<<<(E2 * HEADS + 255) / 256, 256, 0, stream>>>(
        ei, a_s, a_d, m, den, E, E2);
    gat_edge_scatter_kernel<<<(E2 + 1) / 2, 256, 0, stream>>>(
        ei, a_s, a_d, m, den, xh, out, E, E2);
    // head mean already folded (0.25); add bias
    gat_bias_kernel<<<(N * HID + 255) / 256, 256, 0, stream>>>(out, bias, N * HID);

    (void)n_in; (void)out_size; (void)ws_size;
}
